// Transformer_2370821947914
// MI455X (gfx1250) — matrix operand order repaired and checked by reading, not compiled, not run on hardware
//
#include <hip/hip_runtime.h>
#include <hip/hip_bf16.h>

// MI455X / gfx1250 transformer forward: all GEMMs (QKV, QK^T, PV, FF1, FF2)
// via v_wmma_f32_16x16x32_f16 (f16 in, f32 accumulate). Shared operands (W tile,
// K/V tiles) staged into LDS with double-buffered GLOBAL_LOAD_ASYNC_TO_LDS_B128
// + s_wait_asynccnt. Softmax/LayerNorm in f32.

typedef __attribute__((ext_vector_type(16))) _Float16 v16h;
typedef __attribute__((ext_vector_type(8)))  _Float16 v8h;
typedef __attribute__((ext_vector_type(8)))  float    v8f;
typedef _Float16 half_t;

#define S_LEN 2048
#define HID   1024
#define NHEAD 8
#define DH    128

static __device__ __forceinline__ v8f wmma16(v16h a, v16h b, v8f c) {
    return __builtin_amdgcn_wmma_f32_16x16x32_f16(false, a, false, b, (short)0, c,
                                                  false, false);
}

// LDS byte address = low 32 bits of the generic pointer (ISA 10.2: LDS aperture
// truncates to addr[31:0], matching the DS/async VDST offset).
static __device__ __forceinline__ unsigned lds_addr(const void* p) {
    return (unsigned)(unsigned long long)p;
}

// One 16B async global->LDS copy per lane (one wave instruction, ASYNCcnt +1).
static __device__ __forceinline__ void async_ld16(unsigned lds_byte,
                                                  const half_t* g) {
    asm volatile("global_load_async_to_lds_b128 %0, %1, off"
                 :: "v"(lds_byte), "v"((unsigned long long)g)
                 : "memory");
}
static __device__ __forceinline__ void wait_async_le2() {
    asm volatile("s_wait_asynccnt 0x2" ::: "memory");
}
static __device__ __forceinline__ void wait_async_le4() {
    asm volatile("s_wait_asynccnt 0x4" ::: "memory");
}

// ---------------------------------------------------------------------------
// x = inputs + positional encoding; also emit f16 copy.
// pe(s,h) = (h even ? sin : cos)(s * 10000^(-2h/1024))
// ---------------------------------------------------------------------------
__global__ void posenc_kernel(const float* __restrict__ in,
                              float* __restrict__ x32, half_t* __restrict__ xh) {
    int idx = blockIdx.x * blockDim.x + threadIdx.x;       // S*H threads
    int s = idx >> 10;
    int h = idx & 1023;
    float e = -2.0f * (float)h / 1024.0f;
    float freq = expf(e * 9.210340371976184f);             // ln(10000)
    float ang = (float)s * freq;
    float pe = (h & 1) ? cosf(ang) : sinf(ang);
    float v = in[idx] + pe;
    x32[idx] = v;
    xh[idx] = (half_t)v;
}

// ---------------------------------------------------------------------------
// Weight packing into WMMA B-fragment layout:
// dst index = ((kt*64 + nt)*32 + lane)*16 + half
//   K = kt*32 + (lane/16)*16 + half ; N = nt*16 + (lane%16)
// ---------------------------------------------------------------------------
__global__ void pack_w_qkv(const float* __restrict__ W, half_t* __restrict__ P) {
    int idx = blockIdx.x * blockDim.x + threadIdx.x;       // HID*HID threads
    int h  = idx & 15;
    int l  = (idx >> 4) & 31;
    int nt = (idx >> 9) & 63;
    int kt = idx >> 15;
    int K = kt * 32 + (l >> 4) * 16 + h;
    int N = nt * 16 + (l & 15);
    int n = N >> 7, d = N & 127;                           // W is [NH][HID][DH]
    P[idx] = (half_t)W[((size_t)n * HID + K) * DH + d];
}

__global__ void pack_w_hh(const float* __restrict__ W, half_t* __restrict__ P) {
    int idx = blockIdx.x * blockDim.x + threadIdx.x;
    int h  = idx & 15;
    int l  = (idx >> 4) & 31;
    int nt = (idx >> 9) & 63;
    int kt = idx >> 15;
    int K = kt * 32 + (l >> 4) * 16 + h;
    int N = nt * 16 + (l & 15);
    P[idx] = (half_t)W[(size_t)K * HID + N];               // W is [HID][HID]
}

// K (for QK^T): B with Kdim=d (128), Ndim=key row (2048). [NH][4][128][32][16]
__global__ void pack_kb(const half_t* __restrict__ Kh, half_t* __restrict__ Kb) {
    int idx = blockIdx.x * blockDim.x + threadIdx.x;       // NH*S*D threads (2M)
    int h  = idx & 15;
    int l  = (idx >> 4) & 31;
    int st = (idx >> 9) & 127;
    int c  = (idx >> 16) & 3;
    int n  = idx >> 18;
    int srow = st * 16 + (l & 15);
    int dcol = c * 32 + (l >> 4) * 16 + h;
    Kb[idx] = Kh[(size_t)srow * HID + n * DH + dcol];
}

// V (for P*V): B with Kdim=key row (2048), Ndim=d (128). [NH][64][8][32][16]
__global__ void pack_vb(const half_t* __restrict__ Vh, half_t* __restrict__ Vb) {
    int idx = blockIdx.x * blockDim.x + threadIdx.x;       // 2M threads
    int h  = idx & 15;
    int l  = (idx >> 4) & 31;
    int dt = (idx >> 9) & 7;
    int kb = (idx >> 12) & 63;
    int n  = idx >> 18;
    int krow = kb * 32 + (l >> 4) * 16 + h;
    int dcol = dt * 16 + (l & 15);
    Vb[idx] = Vh[(size_t)krow * HID + n * DH + dcol];
}

// ---------------------------------------------------------------------------
// WMMA GEMM: OUT[S,1024] = act( (X[S,1024] @ W[1024,1024] + bias) * scale )
// Block tile 128x128, 8 waves, wave = 16 rows x 128 cols, K-steps of 32.
// B tile (8KB, fragment-packed => contiguous) staged in LDS, double buffered,
// via async global->LDS copies. A rows are wave-private, loaded direct.
// ---------------------------------------------------------------------------
template <bool RELU, bool OUT16, bool OUT32>
__global__ __launch_bounds__(256) void gemm_kernel(
    const half_t* __restrict__ X, const half_t* __restrict__ Wp,
    const float* __restrict__ bias, half_t* __restrict__ out16,
    float* __restrict__ out32, float scale) {
    __shared__ half_t bstage[2][4096];                     // 2 x 8KB
    const int tid = threadIdx.x;
    const int wave = tid >> 5;
    const int lane = tid & 31;
    const int g = lane >> 4, ln = lane & 15;
    const int m0 = blockIdx.x * 128 + wave * 16;
    const int n0 = blockIdx.y * 128;

    v8f acc[8] = {};
    const half_t* xrow = X + (size_t)(m0 + ln) * HID;
    const half_t* wtile = Wp + (size_t)(n0 >> 4) * 512;    // + kt*32768 per step

    // prologue: stage tile 0
#pragma unroll
    for (int i = 0; i < 2; ++i) {
        int f = tid + 256 * i;                             // 0..511 16B pieces
        async_ld16(lds_addr(&bstage[0][f * 8]), wtile + f * 8);
    }

    for (int kt = 0; kt < HID / 32; ++kt) {
        // prefetch next tile (wraps; redundant last fetch keeps waits uniform)
        const int nk = (kt + 1) & 31;
        const int nbuf = (kt + 1) & 1;
#pragma unroll
        for (int i = 0; i < 2; ++i) {
            int f = tid + 256 * i;
            async_ld16(lds_addr(&bstage[nbuf][f * 8]),
                       wtile + (size_t)nk * 32768 + f * 8);
        }
        wait_async_le2();                                  // current tile landed
        __syncthreads();

        union { v16h v; v8h h[2]; } ua;
        ua.h[0] = *(const v8h*)(xrow + kt * 32 + g * 16);
        ua.h[1] = *(const v8h*)(xrow + kt * 32 + g * 16 + 8);
        v16h a = ua.v;
        const half_t* bs = &bstage[kt & 1][0];
#pragma unroll
        for (int nt = 0; nt < 8; ++nt) {
            v16h b = *(const v16h*)(bs + nt * 512 + lane * 16);
            acc[nt] = wmma16(a, b, acc[nt]);
        }
        __syncthreads();                                   // reads done before overwrite
    }
#pragma unroll
    for (int nt = 0; nt < 8; ++nt) {
        int col = n0 + nt * 16 + ln;
        float bv = bias[col];
#pragma unroll
        for (int v = 0; v < 8; ++v) {
            int row = m0 + v + g * 8;                      // C layout: M=v (+8 hi group)
            float val = (acc[nt][v] + bv) * scale;
            if (RELU) val = fmaxf(val, 0.0f);
            if (OUT32) out32[(size_t)row * HID + col] = val;
            if (OUT16) out16[(size_t)row * HID + col] = (half_t)val;
        }
    }
}

// ---------------------------------------------------------------------------
// Flash attention. Grid (S/128, NH), 8 waves; wave = 16 query rows x D=128.
// K/V tiles (8KB each per 32-key block) staged in LDS, double buffered, async.
// Q already carries the 1/sqrt(H) scale.
// ---------------------------------------------------------------------------
__global__ __launch_bounds__(256) void attn_kernel(
    const half_t* __restrict__ Qh, const half_t* __restrict__ Kb,
    const half_t* __restrict__ Vb, float* __restrict__ cat) {
    __shared__ half_t kstage[2][4096];                     // 2 x 8KB
    __shared__ half_t vstage[2][4096];                     // 2 x 8KB
    __shared__ half_t pbuf[8 * 16 * 32];                   // per-wave 16x32 P tile
    const int head = blockIdx.y;
    const int tid = threadIdx.x;
    const int wave = tid >> 5;
    const int lane = tid & 31;
    const int g = lane >> 4, ln = lane & 15;
    const int qbase = blockIdx.x * 128 + wave * 16;
    half_t* myp = pbuf + wave * (16 * 32);

    const half_t* kbase = Kb + (size_t)head * 262144;      // [4][128][32][16]
    const half_t* vbase = Vb + (size_t)head * 262144;      // [64][8][32][16]

    // Hoisted Q A-fragments (row = qbase+ln, 4 chunks over D=128)
    v16h qa[4];
    {
        const half_t* qrow = Qh + (size_t)(qbase + ln) * HID + head * DH;
#pragma unroll
        for (int c = 0; c < 4; ++c) {
            union { v16h v; v8h h[2]; } u;
            u.h[0] = *(const v8h*)(qrow + c * 32 + g * 16);
            u.h[1] = *(const v8h*)(qrow + c * 32 + g * 16 + 8);
            qa[c] = u.v;
        }
    }

    v8f acc[8] = {};
    float mrow[8], lrow[8];
#pragma unroll
    for (int v = 0; v < 8; ++v) { mrow[v] = -1e30f; lrow[v] = 0.0f; }

    // tile stage helper (4 async instructions per wave: 2 for K, 2 for V)
    auto stage_tile = [&](int kb, int buf) {
#pragma unroll
        for (int i = 0; i < 2; ++i) {                      // K: 4 chunks of 2KB
            int f = tid + 256 * i;                         // 0..511
            int cc = f >> 7, w = f & 127;
            async_ld16(lds_addr(&kstage[buf][f * 8]),
                       kbase + (size_t)cc * 65536 + kb * 1024 + w * 8);
        }
#pragma unroll
        for (int i = 0; i < 2; ++i) {                      // V: contiguous 8KB
            int w = tid + 256 * i;                         // 0..511
            async_ld16(lds_addr(&vstage[buf][w * 8]),
                       vbase + (size_t)kb * 4096 + w * 8);
        }
    };

    stage_tile(0, 0);                                      // prologue

    for (int kb = 0; kb < S_LEN / 32; ++kb) {
        stage_tile((kb + 1) & 63, (kb + 1) & 1);           // prefetch next
        wait_async_le4();                                  // current K/V landed
        __syncthreads();
        const half_t* ks = &kstage[kb & 1][0];
        const half_t* vs = &vstage[kb & 1][0];

        // ---- scores: 16 queries x 32 keys = two 16x16 C tiles ----
        v8f sc[2];
#pragma unroll
        for (int t = 0; t < 2; ++t) {
            v8f c = {};
#pragma unroll
            for (int cc = 0; cc < 4; ++cc) {
                v16h b = *(const v16h*)(ks + cc * 1024 + t * 512 + lane * 16);
                c = wmma16(qa[cc], b, c);
            }
            sc[t] = c;
        }
        // ---- online softmax (rows live in 16-lane groups) ----
        float mnew[8], sca[8];
#pragma unroll
        for (int v = 0; v < 8; ++v) {
            float bm = fmaxf(sc[0][v], sc[1][v]);
            bm = fmaxf(bm, __shfl_xor(bm, 1, 32));
            bm = fmaxf(bm, __shfl_xor(bm, 2, 32));
            bm = fmaxf(bm, __shfl_xor(bm, 4, 32));
            bm = fmaxf(bm, __shfl_xor(bm, 8, 32));
            float mn = fmaxf(mrow[v], bm);
            sca[v] = __expf(mrow[v] - mn);
            mnew[v] = mn;
            mrow[v] = mn;
        }
#pragma unroll
        for (int v = 0; v < 8; ++v) {
            float p0 = __expf(sc[0][v] - mnew[v]);
            float p1 = __expf(sc[1][v] - mnew[v]);
            int row = v + g * 8;
            myp[row * 32 + ln] = (half_t)p0;
            myp[row * 32 + 16 + ln] = (half_t)p1;
            float rs = p0 + p1;
            rs += __shfl_xor(rs, 1, 32);
            rs += __shfl_xor(rs, 2, 32);
            rs += __shfl_xor(rs, 4, 32);
            rs += __shfl_xor(rs, 8, 32);
            lrow[v] = lrow[v] * sca[v] + rs;
        }
#pragma unroll
        for (int d = 0; d < 8; ++d)
#pragma unroll
            for (int v = 0; v < 8; ++v) acc[d][v] *= sca[v];
        __syncthreads();                                   // P visible for A-frag read
        // ---- reload P as A-fragment, P @ V ----
        union { v16h v; v8h h[2]; } up;
        up.h[0] = *(const v8h*)(myp + ln * 32 + g * 16);
        up.h[1] = *(const v8h*)(myp + ln * 32 + g * 16 + 8);
        v16h pa = up.v;
#pragma unroll
        for (int d = 0; d < 8; ++d) {
            v16h b = *(const v16h*)(vs + d * 512 + lane * 16);
            acc[d] = wmma16(pa, b, acc[d]);
        }
        __syncthreads();                                   // stage+pbuf reads done
    }
    // ---- normalize and write concat-head output ----
#pragma unroll
    for (int v = 0; v < 8; ++v) {
        int row = qbase + v + g * 8;
        float inv = 1.0f / lrow[v];
#pragma unroll
        for (int d = 0; d < 8; ++d)
            cat[(size_t)row * HID + head * DH + d * 16 + ln] = acc[d][v] * inv;
    }
}

// ---------------------------------------------------------------------------
// out = LayerNorm(A + B); emit f32 and f16. One block per row.
// ---------------------------------------------------------------------------
__global__ __launch_bounds__(256) void addln_kernel(
    const float* __restrict__ A, const float* __restrict__ B,
    float* __restrict__ out32, half_t* __restrict__ out16) {
    __shared__ float s1[256], s2[256];
    int row = blockIdx.x;
    const float* a = A + (size_t)row * HID;
    const float* b = B + (size_t)row * HID;
    float x[4], sum = 0.0f, sq = 0.0f;
#pragma unroll
    for (int i = 0; i < 4; ++i) {
        int c = threadIdx.x + i * 256;
        float v = a[c] + b[c];
        x[i] = v; sum += v; sq += v * v;
    }
    s1[threadIdx.x] = sum; s2[threadIdx.x] = sq;
    __syncthreads();
    for (int off = 128; off > 0; off >>= 1) {
        if (threadIdx.x < off) {
            s1[threadIdx.x] += s1[threadIdx.x + off];
            s2[threadIdx.x] += s2[threadIdx.x + off];
        }
        __syncthreads();
    }
    float mean = s1[0] * (1.0f / HID);
    float var = s2[0] * (1.0f / HID) - mean * mean;
    float inv = rsqrtf(var + 1e-5f);
#pragma unroll
    for (int i = 0; i < 4; ++i) {
        int c = threadIdx.x + i * 256;
        float v = (x[i] - mean) * inv;
        out32[(size_t)row * HID + c] = v;
        out16[(size_t)row * HID + c] = (half_t)v;
    }
}

// ---------------------------------------------------------------------------
extern "C" void kernel_launch(void* const* d_in, const int* in_sizes, int n_in,
                              void* d_out, int out_size, void* d_ws, size_t ws_size,
                              hipStream_t stream) {
    (void)in_sizes; (void)n_in; (void)out_size; (void)ws_size;
    const float* inp = (const float*)d_in[0];
    const float* Wq = (const float*)d_in[1];
    const float* bq = (const float*)d_in[2];
    const float* Wk = (const float*)d_in[3];
    const float* bk = (const float*)d_in[4];
    const float* Wv = (const float*)d_in[5];
    const float* bv = (const float*)d_in[6];
    const float* W1 = (const float*)d_in[7];
    const float* b1 = (const float*)d_in[8];
    const float* W2 = (const float*)d_in[9];
    const float* b2 = (const float*)d_in[10];

    char* ws = (char*)d_ws;
    size_t off = 0;
    auto alloc = [&](size_t bytes) -> void* {
        void* p = ws + off;
        off += (bytes + 255) & ~(size_t)255;
        return p;
    };
    const size_t SH4 = (size_t)S_LEN * HID * 4;
    const size_t SH2 = (size_t)S_LEN * HID * 2;
    const size_t HH2 = (size_t)HID * HID * 2;
    float*  t32   = (float*)alloc(SH4);
    half_t* th    = (half_t*)alloc(SH2);
    float*  mh32  = (float*)alloc(SH4);
    half_t* mh16  = (half_t*)alloc(SH2);
    float*  cat32 = (float*)alloc(SH4);
    float*  ff32  = (float*)alloc(SH4);
    half_t* Kh    = (half_t*)alloc(SH2);
    half_t* Vh    = (half_t*)alloc(SH2);
    half_t* Qh    = (half_t*)alloc(SH2);
    half_t* y1h   = (half_t*)alloc(SH2);
    half_t* KbP   = (half_t*)alloc(SH2);
    half_t* VbP   = (half_t*)alloc(SH2);
    half_t* WqP   = (half_t*)alloc(HH2);
    half_t* WkP   = (half_t*)alloc(HH2);
    half_t* WvP   = (half_t*)alloc(HH2);
    half_t* W1P   = (half_t*)alloc(HH2);
    half_t* W2P   = (half_t*)alloc(HH2);

    const dim3 blk(256);
    const dim3 gemmGrid(S_LEN / 128, HID / 128);
    const int elemsSH = S_LEN * HID;       // 2M
    const int elemsHH = HID * HID;         // 1M

    // x = inputs + posenc
    posenc_kernel<<<elemsSH / 256, blk, 0, stream>>>(inp, t32, th);
    // pack weights to B-fragment layout (f16)
    pack_w_qkv<<<elemsHH / 256, blk, 0, stream>>>(Wq, WqP);
    pack_w_qkv<<<elemsHH / 256, blk, 0, stream>>>(Wk, WkP);
    pack_w_qkv<<<elemsHH / 256, blk, 0, stream>>>(Wv, WvP);
    pack_w_hh<<<elemsHH / 256, blk, 0, stream>>>(W1, W1P);
    pack_w_hh<<<elemsHH / 256, blk, 0, stream>>>(W2, W2P);
    // K/V computed once from layer-0 input (reference semantics)
    gemm_kernel<false, true, false><<<gemmGrid, blk, 0, stream>>>(
        th, WkP, bk, Kh, nullptr, 1.0f);
    gemm_kernel<false, true, false><<<gemmGrid, blk, 0, stream>>>(
        th, WvP, bv, Vh, nullptr, 1.0f);
    pack_kb<<<elemsSH / 256, blk, 0, stream>>>(Kh, KbP);
    pack_vb<<<elemsSH / 256, blk, 0, stream>>>(Vh, VbP);

    const float qscale = 1.0f / 32.0f;     // 1/sqrt(H=1024)
    for (int layer = 0; layer < 6; ++layer) {
        gemm_kernel<false, true, false><<<gemmGrid, blk, 0, stream>>>(
            th, WqP, bq, Qh, nullptr, qscale);
        attn_kernel<<<dim3(S_LEN / 128, NHEAD), blk, 0, stream>>>(Qh, KbP, VbP, cat32);
        addln_kernel<<<S_LEN, blk, 0, stream>>>(t32, cat32, mh32, mh16);
        gemm_kernel<true, true, false><<<gemmGrid, blk, 0, stream>>>(
            mh16, W1P, b1, y1h, nullptr, 1.0f);
        gemm_kernel<false, false, true><<<gemmGrid, blk, 0, stream>>>(
            y1h, W2P, b2, nullptr, ff32, 1.0f);
        float* outp = (layer == 5) ? (float*)d_out : t32;
        addln_kernel<<<S_LEN, blk, 0, stream>>>(mh32, ff32, outp, th);
    }
}
